// DGL_Net_9345848836096
// MI455X (gfx1250) — matrix operand order repaired and checked by reading, not compiled, not run on hardware
//
#include <hip/hip_runtime.h>
#include <hip/hip_bf16.h>

typedef __attribute__((ext_vector_type(16))) _Float16 v16h;
typedef __attribute__((ext_vector_type(2)))  __fp16   v2fp16;   // cvt_pkrtz return type
typedef __attribute__((ext_vector_type(8)))  float    v8f;

#define N_NODES 50000
#define N_EDGES 800000

// ---------------------------------------------------------------- utilities
__global__ void fill_zero(float* __restrict__ p, long long n) {
    long long i = (long long)blockIdx.x * blockDim.x + threadIdx.x;
    if (i < n) p[i] = 0.0f;
}

__global__ void degree_kernel(const int* __restrict__ src, const int* __restrict__ dst,
                              float* __restrict__ deg_out, float* __restrict__ deg_in, int E) {
    int e = blockIdx.x * blockDim.x + threadIdx.x;
    if (e < E) {
        atomicAdd(&deg_out[src[e]], 1.0f);
        atomicAdd(&deg_in[dst[e]], 1.0f);
    }
}

__global__ void rsqrt_clamp(float* __restrict__ p, int n) {
    int i = blockIdx.x * blockDim.x + threadIdx.x;
    if (i < n) p[i] = rsqrtf(fmaxf(p[i], 1.0f));
}

// ------------------------------------------------- WMMA GEMM: C = (A*scale) @ B
// A: MxK f32 (row-scaled by scale[row]), B: KxN f32, C: MxN f32.
// 256 threads = 8 wave32 waves; block tile 128x16, wave tile 16x16, K step 32.
// Double-buffered LDS (packed f16 dwords, 19-dword row stride = conflict-free),
// branch-free clamped global loads, v_cvt_pk_rtz_f16_f32 packing.
__global__ __launch_bounds__(256) void gemm_wmma_f16(
    const float* __restrict__ A, const float* __restrict__ scale,
    const float* __restrict__ B, float* __restrict__ C,
    int M, int N, int K)
{
    __shared__ unsigned int As[2][128][19];  // [buf][row][k-pair]
    __shared__ unsigned int Bs[2][16][19];   // [buf][col][k-pair] (transposed)

    const int tid  = threadIdx.x;
    const int lane = tid & 31;
    const int wave = tid >> 5;
    const int hf   = lane >> 4;   // wave32 lane half
    const int mn   = lane & 15;

    const int rowBase = blockIdx.y * 128;
    const int colBase = blockIdx.x * 16;
    const int tileRow = wave * 16;

    union H2U { v2fp16 h; unsigned int u; };

    unsigned int ra[8];   // staged A tile regs (8 packed pairs / thread)
    unsigned int rb;      // staged B tile reg  (1 packed pair / thread)
    v8f acc = {};

    // branch-free tile load into registers (zeros outside M/N/K via cndmask)
    auto load_tile = [&](int kBase) {
        #pragma unroll
        for (int j = 0; j < 8; ++j) {
            int idx = j * 256 + tid;
            int r = idx >> 4, kp = idx & 15;
            int row = rowBase + r;
            int kk  = kBase + 2 * kp;
            int rowc = min(row, M - 1);
            long long base = (long long)rowc * K;
            if (j == 0) __builtin_prefetch(&A[base + min(kk + 64, K - 1)], 0, 1);
            float sc = scale[rowc];
            float v0 = A[base + min(kk,     K - 1)];
            float v1 = A[base + min(kk + 1, K - 1)];
            v0 = (row < M && kk     < K) ? v0 * sc : 0.0f;
            v1 = (row < M && kk + 1 < K) ? v1 * sc : 0.0f;
            H2U t; t.h = __builtin_amdgcn_cvt_pkrtz(v0, v1);
            ra[j] = t.u;
        }
        {
            int c = tid & 15, kp = tid >> 4;
            int col = colBase + c;
            int kk  = kBase + 2 * kp;
            int colc = min(col, N - 1);
            float v0 = B[(long long)min(kk,     K - 1) * N + colc];
            float v1 = B[(long long)min(kk + 1, K - 1) * N + colc];
            v0 = (col < N && kk     < K) ? v0 : 0.0f;
            v1 = (col < N && kk + 1 < K) ? v1 : 0.0f;
            H2U t; t.h = __builtin_amdgcn_cvt_pkrtz(v0, v1);
            rb = t.u;
        }
    };

    auto store_tile = [&](int buf) {
        #pragma unroll
        for (int j = 0; j < 8; ++j) {
            int idx = j * 256 + tid;
            As[buf][idx >> 4][idx & 15] = ra[j];
        }
        Bs[buf][tid & 15][tid >> 4] = rb;
    };

    auto compute = [&](int buf) {
        union { v16h h; unsigned int u[8]; } ua, ub;
        const unsigned int* arow = &As[buf][tileRow + mn][0];
        const unsigned int* brow = &Bs[buf][mn][0];
        #pragma unroll
        for (int v = 0; v < 8; ++v) {
            ua.u[v] = arow[((v < 4) ? 0 : 8) + hf * 4 + (v & 3)]; // A: K pairs
            ub.u[v] = brow[((v < 4) ? 0 : 8) + hf * 4 + (v & 3)]; // B: K pairs
        }
        acc = __builtin_amdgcn_wmma_f32_16x16x32_f16(
            false, ua.h, false, ub.h, (short)0, acc, false, false);
    };

    // prologue: stage tile 0
    load_tile(0);
    store_tile(0);
    int cur = 0;
    for (int kBase = 0; kBase < K; kBase += 32) {
        load_tile(kBase + 32);      // overlap next-tile loads with compute
        __syncthreads();            // tile `cur` visible / prev readers done
        compute(cur);
        store_tile(cur ^ 1);
        cur ^= 1;
    }

    // store D: VGPR r -> row (r + 8*half), col = lane&15
    const int col = colBase + mn;
    #pragma unroll
    for (int r = 0; r < 8; ++r) {
        int row = rowBase + tileRow + hf * 8 + r;
        if (row < M && col < N) C[(long long)row * N + col] = acc[r];
    }
}

// ---------------------------------------------------- SpMM scatter (L2 atomics)
__global__ void spmm_scatter(const float* __restrict__ h, const int* __restrict__ src,
                             const int* __restrict__ dst, float* __restrict__ agg,
                             int E, int F) {
    long long t = (long long)blockIdx.x * blockDim.x + threadIdx.x;
    if (t >= (long long)E * F) return;
    int e = (int)(t / F);
    int f = (int)(t - (long long)e * F);
    atomicAdd(&agg[(long long)dst[e] * F + f], h[(long long)src[e] * F + f]);
}

// ---------------------------------------------------------------- epilogues
__global__ void epilogue_relu(const float* __restrict__ agg, const float* __restrict__ cdst,
                              const float* __restrict__ bias, float* __restrict__ out,
                              int M, int F) {
    long long t = (long long)blockIdx.x * blockDim.x + threadIdx.x;
    if (t >= (long long)M * F) return;
    int row = (int)(t / F);
    int col = (int)(t - (long long)row * F);
    out[t] = fmaxf(agg[t] * cdst[row] + bias[col], 0.0f);
}

__global__ void epilogue_logsoftmax7(const float* __restrict__ agg, const float* __restrict__ cdst,
                                     const float* __restrict__ bias, float* __restrict__ out, int M) {
    int row = blockIdx.x * blockDim.x + threadIdx.x;
    if (row >= M) return;
    float c = cdst[row];
    float z[7];
    float mx = -1e30f;
    #pragma unroll
    for (int j = 0; j < 7; ++j) { z[j] = agg[row * 7 + j] * c + bias[j]; mx = fmaxf(mx, z[j]); }
    float s = 0.0f;
    #pragma unroll
    for (int j = 0; j < 7; ++j) s += __expf(z[j] - mx);
    float lse = mx + __logf(s);
    #pragma unroll
    for (int j = 0; j < 7; ++j) out[row * 7 + j] = z[j] - lse;
}

// ---------------------------------------------------------------------------
static inline int cdiv(long long a, long long b) { return (int)((a + b - 1) / b); }

extern "C" void kernel_launch(void* const* d_in, const int* in_sizes, int n_in,
                              void* d_out, int out_size, void* d_ws, size_t ws_size,
                              hipStream_t stream) {
    const float* features = (const float*)d_in[0];
    const int*   edge     = (const int*)d_in[1];   // (2, E): src then dst
    const float* W1 = (const float*)d_in[2]; const float* b1 = (const float*)d_in[3];
    const float* W2 = (const float*)d_in[4]; const float* b2 = (const float*)d_in[5];
    const float* W3 = (const float*)d_in[6]; const float* b3 = (const float*)d_in[7];
    float* out = (float*)d_out;

    const int* src = edge;
    const int* dst = edge + N_EDGES;

    float* ws    = (float*)d_ws;
    float* c_src = ws;                                  // [N_NODES]
    float* c_dst = ws + N_NODES;                        // [N_NODES]
    float* bufA  = ws + 2 * N_NODES;                    // [N_NODES*256]
    float* bufB  = bufA + (long long)N_NODES * 256;     // [N_NODES*256]

    // normalization coefficients
    fill_zero<<<cdiv(2LL * N_NODES, 256), 256, 0, stream>>>(c_src, 2LL * N_NODES);
    degree_kernel<<<cdiv(N_EDGES, 256), 256, 0, stream>>>(src, dst, c_src, c_dst, N_EDGES);
    rsqrt_clamp<<<cdiv(2LL * N_NODES, 256), 256, 0, stream>>>(c_src, 2 * N_NODES);

    // ---- layer 1: 1433 -> 256
    gemm_wmma_f16<<<dim3(cdiv(256, 16), cdiv(N_NODES, 128)), 256, 0, stream>>>(
        features, c_src, W1, bufA, N_NODES, 256, 1433);
    fill_zero<<<cdiv((long long)N_NODES * 256, 256), 256, 0, stream>>>(bufB, (long long)N_NODES * 256);
    spmm_scatter<<<cdiv((long long)N_EDGES * 256, 256), 256, 0, stream>>>(bufA, src, dst, bufB, N_EDGES, 256);
    epilogue_relu<<<cdiv((long long)N_NODES * 256, 256), 256, 0, stream>>>(bufB, c_dst, b1, bufA, N_NODES, 256);

    // ---- layer 2: 256 -> 32   (x1 in bufA)
    gemm_wmma_f16<<<dim3(cdiv(32, 16), cdiv(N_NODES, 128)), 256, 0, stream>>>(
        bufA, c_src, W2, bufB, N_NODES, 32, 256);
    fill_zero<<<cdiv((long long)N_NODES * 32, 256), 256, 0, stream>>>(bufA, (long long)N_NODES * 32);
    spmm_scatter<<<cdiv((long long)N_EDGES * 32, 256), 256, 0, stream>>>(bufB, src, dst, bufA, N_EDGES, 32);
    epilogue_relu<<<cdiv((long long)N_NODES * 32, 256), 256, 0, stream>>>(bufA, c_dst, b2, bufB, N_NODES, 32);

    // ---- layer 3: 32 -> 7     (x2 in bufB)
    gemm_wmma_f16<<<dim3(cdiv(7, 16), cdiv(N_NODES, 128)), 256, 0, stream>>>(
        bufB, c_src, W3, bufA, N_NODES, 7, 32);
    fill_zero<<<cdiv((long long)N_NODES * 7, 256), 256, 0, stream>>>(bufB, (long long)N_NODES * 7);
    spmm_scatter<<<cdiv((long long)N_EDGES * 7, 256), 256, 0, stream>>>(bufA, src, dst, bufB, N_EDGES, 7);
    epilogue_logsoftmax7<<<cdiv(N_NODES, 256), 256, 0, stream>>>(bufB, c_dst, b3, out, N_NODES);
}
